// FullAttention_73323681677865
// MI455X (gfx1250) — matrix operand order repaired and checked by reading, not compiled, not run on hardware
//
#include <hip/hip_runtime.h>
#include <type_traits>

typedef __attribute__((ext_vector_type(16))) _Float16 v16h;
typedef __attribute__((ext_vector_type(8)))  float    v8f;
typedef __attribute__((ext_vector_type(2)))  _Float16 h2;

namespace {
constexpr int kB = 4, kL = 2048, kH = 16, kD = 64;
constexpr int BLOCK_M  = 64;              // query rows per workgroup (4 waves x 16)
constexpr int BLOCK_N  = 64;              // key rows per iteration
constexpr int NWAVES   = BLOCK_M / 16;
constexpr int NTHREADS = NWAVES * 32;
constexpr int KPAD     = 4;               // halves of padding per LDS row
// softmax in base-2 domain: softmax(scale*x) == 2^(scale*log2e*x) normalized
constexpr float kScaleL2E  = 0.125f * 1.44269504088896340736f;   // 1/sqrt(64)*log2(e)
constexpr float kMaskedL2E = -1.0e9f * 0.125f * 1.44269504088896340736f;
}

static __device__ __forceinline__ float fast_exp2(float x) {
#if __has_builtin(__builtin_amdgcn_exp2f)
  return __builtin_amdgcn_exp2f(x);
#else
  return exp2f(x);
#endif
}

static __device__ __forceinline__ float fast_rcp(float x) {
#if __has_builtin(__builtin_amdgcn_rcpf)
  return __builtin_amdgcn_rcpf(x);
#else
  return 1.0f / x;
#endif
}

// xor-butterfly within 16-lane group via ds_swizzle (group-of-32 encoding:
// offset15=0, {xor[14:10], or[9:5], and[4:0]})
template <int MASK>
static __device__ __forceinline__ float swz_xor(float x) {
#if __has_builtin(__builtin_amdgcn_ds_swizzle)
  return __builtin_bit_cast(
      float, __builtin_amdgcn_ds_swizzle(__builtin_bit_cast(int, x),
                                         (MASK << 10) | 0x1f));
#else
  return __shfl_xor(x, MASK, 32);
#endif
}

static __device__ __forceinline__ float red_max16(float x) {
  x = fmaxf(x, swz_xor<1>(x));
  x = fmaxf(x, swz_xor<2>(x));
  x = fmaxf(x, swz_xor<4>(x));
  x = fmaxf(x, swz_xor<8>(x));
  return x;
}

static __device__ __forceinline__ float red_sum16(float x) {
  x += swz_xor<1>(x);
  x += swz_xor<2>(x);
  x += swz_xor<4>(x);
  x += swz_xor<8>(x);
  return x;
}

// A-matrix (16x32 f16) K-offset for vgpr-pair v in lane-half h (ISA 7.12.2)
static __device__ __forceinline__ int kbaseA(int v, int h) {
  return (v < 4) ? (8 * h + 2 * v) : (16 + 8 * h + 2 * (v - 4));
}

__global__ __launch_bounds__(NTHREADS)
void fa_fwd_causal(const float* __restrict__ Q, const float* __restrict__ K,
                   const float* __restrict__ V, float* __restrict__ O) {
  __shared__ _Float16 Ks[BLOCK_N][kD + KPAD];       // keys x d, row-major
  __shared__ _Float16 Vt[kD][BLOCK_N + KPAD];       // TRANSPOSED: d x keys
  __shared__ _Float16 Ps[NWAVES][16][BLOCK_N + 2];  // wave-private P bounce

  const int tid   = threadIdx.x;
  const int wave  = tid >> 5;
  const int lane  = tid & 31;
  const int n16   = lane & 15;   // N index within 16-wide group
  const int lhalf = lane >> 4;   // which half of the wave

  const int mblk = blockIdx.x * BLOCK_M;
  const int bh   = blockIdx.y;
  const int b    = bh >> 4;         // / kH
  const int hd   = bh & (kH - 1);   // % kH
  const int m0   = mblk + wave * 16;

  const size_t rowstride = (size_t)kH * kD;   // elements per sequence position
  const size_t base_bh   = (size_t)b * kL * rowstride + (size_t)hd * kD;

  // ---- Q fragment: rows m0..m0+15, D=64 -> two 16x32 f16 A-tiles ----
  v16h qa[2];
  {
    const float* qrow = Q + base_bh + (size_t)(m0 + n16) * rowstride;
    #pragma unroll
    for (int u = 0; u < 2; ++u) {
      #pragma unroll
      for (int v = 0; v < 8; ++v) {
        const int d = 32 * u + kbaseA(v, lhalf);
        const float2 q2 = *reinterpret_cast<const float2*>(qrow + d);
        qa[u][2 * v]     = (_Float16)q2.x;
        qa[u][2 * v + 1] = (_Float16)q2.y;
      }
    }
  }

  v8f oacc[4];
  #pragma unroll
  for (int t = 0; t < 4; ++t)
    #pragma unroll
    for (int i = 0; i < 8; ++i) oacc[t][i] = 0.0f;

  float m_i[8], l_i[8];
  #pragma unroll
  for (int v = 0; v < 8; ++v) { m_i[v] = -3.0e38f; l_i[v] = 0.0f; }

  const int j_end = mblk + BLOCK_M;   // causal bound for this block

  for (int j = 0; j < j_end; j += BLOCK_N) {
    const bool more = (j + BLOCK_N) < j_end;
    // ---- cooperative stage: K row-major, V transposed (fp32 -> f16) ----
    #pragma unroll
    for (int c = 0; c < (BLOCK_N * kD) / (4 * NTHREADS); ++c) {
      const int f   = tid + c * NTHREADS;   // float4 index within 64x64 tile
      const int r   = f >> 4;               // key row
      const int col = (f & 15) << 2;        // d column
      const size_t goff = base_bh + (size_t)(j + r) * rowstride + col;
      const float4 k4 = *reinterpret_cast<const float4*>(K + goff);
      const float4 v4 = *reinterpret_cast<const float4*>(V + goff);
      Ks[r][col + 0] = (_Float16)k4.x;  Ks[r][col + 1] = (_Float16)k4.y;
      Ks[r][col + 2] = (_Float16)k4.z;  Ks[r][col + 3] = (_Float16)k4.w;
      Vt[col + 0][r] = (_Float16)v4.x;  Vt[col + 1][r] = (_Float16)v4.y;
      Vt[col + 2][r] = (_Float16)v4.z;  Vt[col + 3][r] = (_Float16)v4.w;
      if (more) {  // pull next tile toward caches (global_prefetch_b8)
        const size_t noff = goff + (size_t)BLOCK_N * rowstride;
        __builtin_prefetch(K + noff, 0, 1);
        __builtin_prefetch(V + noff, 0, 1);
      }
    }
    __syncthreads();

    // ---- S = Q @ K^T : four 16x16 f32 tiles, B-fragments pipelined ----
    auto load_kf = [&](int t, int u) {
      v16h kf;  // B-tile: (32 d-contraction) x (16 keys), keys on lanes
      #pragma unroll
      for (int v = 0; v < 8; ++v) {
        const h2 kk = *reinterpret_cast<const h2*>(
            &Ks[16 * t + n16][32 * u + kbaseA(v, lhalf)]);
        kf[2 * v]     = kk.x;
        kf[2 * v + 1] = kk.y;
      }
      return kf;
    };
    v8f s[4];
    #pragma unroll
    for (int t = 0; t < 4; ++t)
      #pragma unroll
      for (int i = 0; i < 8; ++i) s[t][i] = 0.0f;
    {
      v16h kfa = load_kf(0, 0), kfb = load_kf(0, 1);
      #pragma unroll
      for (int t = 0; t < 4; ++t) {
        v16h nka = kfa, nkb = kfb;
        if (t < 3) { nka = load_kf(t + 1, 0); nkb = load_kf(t + 1, 1); }
        s[t] = __builtin_amdgcn_wmma_f32_16x16x32_f16(false, qa[0], false, kfa,
                                                      (short)0, s[t], false, false);
        s[t] = __builtin_amdgcn_wmma_f32_16x16x32_f16(false, qa[1], false, kfb,
                                                      (short)0, s[t], false, false);
        kfa = nka; kfb = nkb;
      }
    }

    // ---- online softmax update (base-2 domain) ----
    // Mask only needed when this wave's rows intersect the block diagonal.
    auto softmax_body = [&](auto masked_c) {
      constexpr bool MASKED = masked_c.value;
      #pragma unroll
      for (int v = 0; v < 8; ++v) {
        const int row = m0 + v + 8 * lhalf;   // C layout: M = v + 8*(lane/16)
        float x[4];
        #pragma unroll
        for (int t = 0; t < 4; ++t) {
          x[t] = s[t][v] * kScaleL2E;
          if constexpr (MASKED) {
            if (j + 16 * t + n16 > row) x[t] = kMaskedL2E;
          }
        }
        float mx = fmaxf(fmaxf(x[0], x[1]), fmaxf(x[2], x[3]));
        mx = red_max16(mx);
        const float mnew  = fmaxf(m_i[v], mx);
        const float alpha = fast_exp2(m_i[v] - mnew);
        float p[4];
        float rs = 0.0f;
        #pragma unroll
        for (int t = 0; t < 4; ++t) { p[t] = fast_exp2(x[t] - mnew); rs += p[t]; }
        rs = red_sum16(rs);
        l_i[v] = l_i[v] * alpha + rs;
        m_i[v] = mnew;
        #pragma unroll
        for (int t = 0; t < 4; ++t) oacc[t][v] *= alpha;
        #pragma unroll
        for (int t = 0; t < 4; ++t)
          Ps[wave][v + 8 * lhalf][16 * t + n16] = (_Float16)p[t];
      }
    };
    if ((j + BLOCK_N) > m0) {                     // wave-uniform branch
      softmax_body(std::true_type{});
    } else {
      softmax_body(std::false_type{});
    }

    asm volatile("s_wait_dscnt 0x0" ::: "memory");  // wave-private LDS bounce

    // ---- reload P as two 16x32 A-tiles; O += P @ V, pipelined ----
    v16h pf[2];
    #pragma unroll
    for (int u = 0; u < 2; ++u) {
      #pragma unroll
      for (int v = 0; v < 8; ++v) {
        const h2 pp = *reinterpret_cast<const h2*>(
            &Ps[wave][n16][32 * u + kbaseA(v, lhalf)]);
        pf[u][2 * v]     = pp.x;
        pf[u][2 * v + 1] = pp.y;
      }
    }
    auto load_vf = [&](int t, int u) {
      v16h vf;  // B-tile: (32 keys contraction) x (16 d-cols), d on lanes
      #pragma unroll
      for (int v = 0; v < 8; ++v) {
        const h2 vv = *reinterpret_cast<const h2*>(
            &Vt[16 * t + n16][32 * u + kbaseA(v, lhalf)]);
        vf[2 * v]     = vv.x;
        vf[2 * v + 1] = vv.y;
      }
      return vf;
    };
    {
      v16h vfa = load_vf(0, 0), vfb = load_vf(0, 1);
      #pragma unroll
      for (int t = 0; t < 4; ++t) {
        v16h nva = vfa, nvb = vfb;
        if (t < 3) { nva = load_vf(t + 1, 0); nvb = load_vf(t + 1, 1); }
        oacc[t] = __builtin_amdgcn_wmma_f32_16x16x32_f16(false, pf[0], false, vfa,
                                                         (short)0, oacc[t],
                                                         false, false);
        oacc[t] = __builtin_amdgcn_wmma_f32_16x16x32_f16(false, pf[1], false, vfb,
                                                         (short)0, oacc[t],
                                                         false, false);
        vfa = nva; vfb = nvb;
      }
    }
    __syncthreads();
  }

  // ---- epilogue: normalize and store [B, L, H, D] ----
  #pragma unroll
  for (int v = 0; v < 8; ++v) {
    const int row = m0 + v + 8 * lhalf;
    const float inv = fast_rcp(l_i[v]);
    float* orow = O + base_bh + (size_t)row * rowstride;
    #pragma unroll
    for (int t = 0; t < 4; ++t) orow[16 * t + n16] = oacc[t][v] * inv;
  }
}

extern "C" void kernel_launch(void* const* d_in, const int* in_sizes, int n_in,
                              void* d_out, int out_size, void* d_ws, size_t ws_size,
                              hipStream_t stream) {
  const float* Q = (const float*)d_in[0];
  const float* K = (const float*)d_in[1];
  const float* V = (const float*)d_in[2];
  float* O = (float*)d_out;
  (void)in_sizes; (void)n_in; (void)out_size; (void)d_ws; (void)ws_size;
  dim3 grid(kL / BLOCK_M, kB * kH, 1);
  dim3 block(NTHREADS, 1, 1);
  fa_fwd_causal<<<grid, block, 0, stream>>>(Q, K, V, O);
}
